// MatrixAttention_39256001085950
// MI455X (gfx1250) — matrix operand order repaired and checked by reading, not compiled, not run on hardware
//
#include <hip/hip_runtime.h>

typedef __attribute__((ext_vector_type(16))) __bf16 bf16x16;
typedef __attribute__((ext_vector_type(8)))  __bf16 bf16x8;
typedef __attribute__((ext_vector_type(8)))  float  f32x8;

// ---------------------------------------------------------------------------
// WMMA wrapper: D = A(16x32 bf16) * B(32x16 bf16) + C(16x16 f32)
// ---------------------------------------------------------------------------
__device__ __forceinline__ f32x8 wmma_bf16(bf16x16 a, bf16x16 b, f32x8 c) {
  return __builtin_amdgcn_wmma_f32_16x16x32_bf16(false, a, false, b, (short)0, c,
                                                 false, false);
}

// A-fragment (16x32, row-major source with 'stride' elems per row).
// Per ISA 7.12.2: lane L holds row m=L&15, K runs [c0,c0+8) and [c0+16,c0+24),
// c0 = 8*(L>>4).
__device__ __forceinline__ bf16x16 load_a_frag(const __bf16* base, int stride,
                                               int lane) {
  int m  = lane & 15;
  int c0 = (lane >> 4) << 3;
  const __bf16* p = base + (size_t)m * stride + c0;
  bf16x16 r;
  *(bf16x8*)&r       = *(const bf16x8*)(p);
  *((bf16x8*)&r + 1) = *(const bf16x8*)(p + 16);
  return r;
}

// B-fragment (32x16, K x N) from N-major storage baseT[n*stride + k]
// (i.e. B-transposed row-major). Lane L: col n=L&15, K in [8g,8g+8) and [8g+16,8g+24), g=L>>4.
__device__ __forceinline__ bf16x16 load_b_frag(const __bf16* baseT, int stride,
                                               int lane) {
  int n  = lane & 15;
  int k0 = (lane >> 4) << 3;
  const __bf16* p = baseT + (size_t)n * stride + k0;
  bf16x16 r;
  *(bf16x8*)&r       = *(const bf16x8*)(p);
  *((bf16x8*)&r + 1) = *(const bf16x8*)(p + 16);
  return r;
}

// ---------------------------------------------------------------------------
// Kernel 0: weight prep — transpose + convert to bf16.
//   Ut[m][n]  = qkv_u[n][m]    (64x64)
//   Vt[k][d]  = qkv_v[d][k]    (192x64)
//   Put[m][n] = proj_u[n][m]   (64x64)
//   Pvt[k][d] = proj_v[d][k]   (64x64)
// ---------------------------------------------------------------------------
__global__ void k0_prep(const float* __restrict__ qkv_u,
                        const float* __restrict__ qkv_v,
                        const float* __restrict__ proj_u,
                        const float* __restrict__ proj_v,
                        __bf16* __restrict__ Ut, __bf16* __restrict__ Vt,
                        __bf16* __restrict__ Put, __bf16* __restrict__ Pvt) {
  int t = blockIdx.x * blockDim.x + threadIdx.x;
  int stride = gridDim.x * blockDim.x;
  for (int idx = t; idx < 64 * 64; idx += stride) {
    int m = idx >> 6, n = idx & 63;
    Ut[idx]  = (__bf16)qkv_u[n * 64 + m];
    Put[idx] = (__bf16)proj_u[n * 64 + m];
    Pvt[idx] = (__bf16)proj_v[n * 64 + m];
  }
  for (int idx = t; idx < 192 * 64; idx += stride) {
    int k = idx >> 6, d = idx & 63;
    Vt[idx] = (__bf16)qkv_v[d * 192 + k];
  }
}

// ---------------------------------------------------------------------------
// Kernel 1: QKV bilinear projection + permuted scatter.
// One wave per (b,l). T = Ut*X (via X^T in LDS), QKV = T*V, scatter with the
// t = 3m + c permutation into head-major staging. Q pre-scaled by 16 (1/SCALE).
//   Qs/Ks: [b*16+head][l][256] bf16
//   Vs   : [b*16+head][f=256][l=2048] bf16  (feature-major = V^T rows)
// ---------------------------------------------------------------------------
#define K1_PITCH 72

__global__ void k1_qkv(const float* __restrict__ x,
                       const __bf16* __restrict__ Ut,
                       const __bf16* __restrict__ Vt,
                       __bf16* __restrict__ Qs, __bf16* __restrict__ Ks,
                       __bf16* __restrict__ Vs) {
  __shared__ __align__(16) __bf16 lds[2][2][64 * K1_PITCH];
  int lane = threadIdx.x & 31;
  int wv   = threadIdx.x >> 5;
  int wid  = blockIdx.x * 2 + wv;   // 0..4095 == b*2048 + l
  int b    = wid >> 11;
  int l    = wid & 2047;

  __bf16* Xt = &lds[wv][0][0];      // X^T: Xt[d][n]
  __bf16* T  = &lds[wv][1][0];      // T  : T[m][d]
  const float* xp = x + (size_t)wid * 4096;

  // Load X (row-major [n][d]) as float4, store transposed to LDS as bf16.
  for (int it = 0; it < 32; ++it) {
    int idx = it * 128 + lane * 4;
    int n = idx >> 6, d = idx & 63;
    float4 vv = *(const float4*)(xp + idx);
    Xt[(d + 0) * K1_PITCH + n] = (__bf16)vv.x;
    Xt[(d + 1) * K1_PITCH + n] = (__bf16)vv.y;
    Xt[(d + 2) * K1_PITCH + n] = (__bf16)vv.z;
    Xt[(d + 3) * K1_PITCH + n] = (__bf16)vv.w;
  }

  // Stage A: T[m][d] = sum_n Ut[m][n] * X[n][d]
#pragma unroll
  for (int mt = 0; mt < 4; ++mt) {
    bf16x16 a0 = load_a_frag(Ut + mt * 16 * 64 + 0,  64, lane);
    bf16x16 a1 = load_a_frag(Ut + mt * 16 * 64 + 32, 64, lane);
#pragma unroll
    for (int dt = 0; dt < 4; ++dt) {
      f32x8 acc = {};
      acc = wmma_bf16(a0, load_b_frag(Xt + dt * 16 * K1_PITCH + 0,  K1_PITCH, lane), acc);
      acc = wmma_bf16(a1, load_b_frag(Xt + dt * 16 * K1_PITCH + 32, K1_PITCH, lane), acc);
      int d  = dt * 16 + (lane & 15);
      int mB = mt * 16 + ((lane >> 4) << 3);
#pragma unroll
      for (int v = 0; v < 8; ++v)
        T[(mB + v) * K1_PITCH + d] = (__bf16)acc[v];
    }
  }

  // Stage B: QKV[m][kk] = sum_d T[m][d] * V[d][kk]; scatter per permutation.
#pragma unroll
  for (int mt = 0; mt < 4; ++mt) {
    bf16x16 a0 = load_a_frag(T + mt * 16 * K1_PITCH + 0,  K1_PITCH, lane);
    bf16x16 a1 = load_a_frag(T + mt * 16 * K1_PITCH + 32, K1_PITCH, lane);
    for (int kt = 0; kt < 12; ++kt) {
      f32x8 acc = {};
      acc = wmma_bf16(a0, load_b_frag(Vt + kt * 16 * 64 + 0,  64, lane), acc);
      acc = wmma_bf16(a1, load_b_frag(Vt + kt * 16 * 64 + 32, 64, lane), acc);
      int kk = kt * 16 + (lane & 15);
      int c = kk >> 6, r = kk & 63;
      int mB = mt * 16 + ((lane >> 4) << 3);
#pragma unroll
      for (int v = 0; v < 8; ++v) {
        int m = mB + v;
        int t = 3 * m + c;
        int s = t >> 6, p = t & 63;
        int head = ((p >> 4) << 2) | (r >> 4);
        int f    = ((p & 15) << 4) | (r & 15);
        size_t bh = (size_t)(b * 16 + head);
        float val = acc[v];
        if (s == 0)      Qs[(bh * 2048 + l) * 256 + f] = (__bf16)(val * 16.0f);
        else if (s == 1) Ks[(bh * 2048 + l) * 256 + f] = (__bf16)val;
        else             Vs[(bh * 256 + f) * 2048 + l] = (__bf16)val;
      }
    }
  }
}

// ---------------------------------------------------------------------------
// Kernel 2: flash attention per head, transposed formulation.
//   S^T = K * Q^T  (A = K rows, B = Q rows as B^T)  -> lane owns row i stats
//   O^T = V^T * P^T (A = Vs rows, B = P via LDS bounce)
// Block: 8 waves, each owns a 16-row Q stripe; grid (2048/128, 32 heads).
// ---------------------------------------------------------------------------
__global__ void k2_attn(const __bf16* __restrict__ Qs,
                        const __bf16* __restrict__ Ks,
                        const __bf16* __restrict__ Vs,
                        __bf16* __restrict__ Yt) {
  __shared__ __align__(16) __bf16 plds[8][16 * 32];
  int lane = threadIdx.x & 31;
  int wv   = threadIdx.x >> 5;
  int bh   = blockIdx.y;                 // b*16 + head
  int b    = bh >> 4, head = bh & 15;
  int mc   = head >> 2, mr = head & 3;
  int q0   = blockIdx.x * 128 + wv * 16; // first Q row of this wave

  const __bf16* Qh = Qs + (size_t)bh * 2048 * 256;
  const __bf16* Kh = Ks + (size_t)bh * 2048 * 256;
  const __bf16* Vh = Vs + (size_t)bh * 256 * 2048;
  __bf16* pw = &plds[wv][0];
  int i = lane & 15;
  int g = lane >> 4;

  // Q fragments (B side of S^T): resident for the whole pass.
  bf16x16 bq[8];
#pragma unroll
  for (int fs = 0; fs < 8; ++fs)
    bq[fs] = load_b_frag(Qh + (size_t)q0 * 256 + fs * 32, 256, lane);

  f32x8 accO[16];
#pragma unroll
  for (int t = 0; t < 16; ++t) accO[t] = (f32x8){};
  float mrow = -3.0e38f, lrow = 0.0f;

  for (int j0 = 0; j0 < 2048; j0 += 32) {
    // Prefetch next K tile (one row per lane) -> global_prefetch_b8.
    if (j0 + 32 < 2048)
      __builtin_prefetch(Kh + (size_t)(j0 + 32 + lane) * 256, 0, 3);

    // S^T tiles: s0 = rows j0..j0+16, s1 = rows j0+16..j0+32 (cols = Q rows i)
    f32x8 s0 = {}, s1 = {};
#pragma unroll
    for (int fs = 0; fs < 8; ++fs) {
      bf16x16 a0 = load_a_frag(Kh + (size_t)j0 * 256 + fs * 32, 256, lane);
      bf16x16 a1 = load_a_frag(Kh + (size_t)(j0 + 16) * 256 + fs * 32, 256, lane);
      s0 = wmma_bf16(a0, bq[fs], s0);
      s1 = wmma_bf16(a1, bq[fs], s1);
    }
    // Lane holds row i of S: j' = v+8g (s0) and 16+v+8g (s1).
    float cur = -3.0e38f;
#pragma unroll
    for (int v = 0; v < 8; ++v) cur = fmaxf(cur, fmaxf(s0[v], s1[v]));
    cur = fmaxf(cur, __shfl_xor(cur, 16, 32));
    float mnew  = fmaxf(mrow, cur);
    float alpha = __expf(mrow - mnew);
    mrow = mnew;

    float e0[8], e1[8], rs = 0.0f;
#pragma unroll
    for (int v = 0; v < 8; ++v) {
      e0[v] = __expf(s0[v] - mnew);
      e1[v] = __expf(s1[v] - mnew);
      rs += e0[v] + e1[v];
    }
    rs += __shfl_xor(rs, 16, 32);
    lrow = lrow * alpha + rs;

    // Rescale previous accumulator only when the running max moved anywhere
    // in the wave (wave-uniform branch; no WMMA inside, EXEC rule untouched).
    if (__ballot(alpha < 1.0f) != 0ull) {
#pragma unroll
      for (int t = 0; t < 16; ++t)
#pragma unroll
        for (int v = 0; v < 8; ++v) accO[t][v] *= alpha;
    }

    // P -> LDS row-major p[i][j'] (bf16), then one B-fragment covers K=32.
#pragma unroll
    for (int v = 0; v < 8; ++v) {
      pw[i * 32 + 8 * g + v]      = (__bf16)e0[v];
      pw[i * 32 + 16 + 8 * g + v] = (__bf16)e1[v];
    }
    bf16x16 bp = load_b_frag(pw, 32, lane);

    // O^T[f][i] += V^T[f][j'] * P^T[j'][i]
#pragma unroll
    for (int ft = 0; ft < 16; ++ft) {
      bf16x16 a = load_a_frag(Vh + (size_t)(ft * 16) * 2048 + j0, 2048, lane);
      accO[ft] = wmma_bf16(a, bp, accO[ft]);
    }
  }

  // Finalize and scatter into Y^T staging: Yt[b][l][r][p] bf16.
  float inv = 1.0f / lrow;
  int lq = q0 + i;
#pragma unroll
  for (int ft = 0; ft < 16; ++ft) {
#pragma unroll
    for (int v = 0; v < 8; ++v) {
      int f  = ft * 16 + 8 * g + v;
      int hc = f >> 4, hr = f & 15;
      int p  = mc * 16 + hc;
      int r  = mr * 16 + hr;
      Yt[(((size_t)(b * 2048 + lq)) * 64 + r) * 64 + p] =
          (__bf16)(accO[ft][v] * inv);
    }
  }
}

// ---------------------------------------------------------------------------
// Kernel 3: output projection OUT = Pu^T * Y * Pv, one wave per (b,l).
// Y read as B-frags from Yt (Y^T layout), T2 bounced through LDS, f32 out.
// ---------------------------------------------------------------------------
__global__ void k3_proj(const __bf16* __restrict__ Yt,
                        const __bf16* __restrict__ Put,
                        const __bf16* __restrict__ Pvt,
                        float* __restrict__ out) {
  __shared__ __align__(16) __bf16 lds3[2][64 * K1_PITCH];
  int lane = threadIdx.x & 31;
  int wv   = threadIdx.x >> 5;
  int wid  = blockIdx.x * 2 + wv;
  __bf16* T = &lds3[wv][0];
  const __bf16* Yl = Yt + (size_t)wid * 4096;  // Y^T[d][n]

  // Stage 1: T2[m][d] = sum_n Put[m][n] * Y[n][d]
#pragma unroll
  for (int mt = 0; mt < 4; ++mt) {
    bf16x16 a0 = load_a_frag(Put + mt * 16 * 64 + 0,  64, lane);
    bf16x16 a1 = load_a_frag(Put + mt * 16 * 64 + 32, 64, lane);
#pragma unroll
    for (int dt = 0; dt < 4; ++dt) {
      f32x8 acc = {};
      acc = wmma_bf16(a0, load_b_frag(Yl + dt * 16 * 64 + 0,  64, lane), acc);
      acc = wmma_bf16(a1, load_b_frag(Yl + dt * 16 * 64 + 32, 64, lane), acc);
      int d  = dt * 16 + (lane & 15);
      int mB = mt * 16 + ((lane >> 4) << 3);
#pragma unroll
      for (int v = 0; v < 8; ++v)
        T[(mB + v) * K1_PITCH + d] = (__bf16)acc[v];
    }
  }

  // Stage 2: OUT[m][k] = sum_d T2[m][d] * Pv[d][k]
  float* op = out + (size_t)wid * 4096;
#pragma unroll
  for (int mt = 0; mt < 4; ++mt) {
    bf16x16 a0 = load_a_frag(T + mt * 16 * K1_PITCH + 0,  K1_PITCH, lane);
    bf16x16 a1 = load_a_frag(T + mt * 16 * K1_PITCH + 32, K1_PITCH, lane);
#pragma unroll
    for (int kt = 0; kt < 4; ++kt) {
      f32x8 acc = {};
      acc = wmma_bf16(a0, load_b_frag(Pvt + kt * 16 * 64 + 0,  64, lane), acc);
      acc = wmma_bf16(a1, load_b_frag(Pvt + kt * 16 * 64 + 32, 64, lane), acc);
      int k  = kt * 16 + (lane & 15);
      int mB = mt * 16 + ((lane >> 4) << 3);
#pragma unroll
      for (int v = 0; v < 8; ++v)
        op[(size_t)(mB + v) * 64 + k] = acc[v];
    }
  }
}

// ---------------------------------------------------------------------------
// Launch. Workspace layout (bytes):
//   [0, 64K): bf16 weights (Ut 8K | Vt 24K | Put 8K | Pvt 8K)
//   then 4 x 32MB: Qs, Ks, Vs, Yt   (total ~128.06 MB)
// ---------------------------------------------------------------------------
extern "C" void kernel_launch(void* const* d_in, const int* in_sizes, int n_in,
                              void* d_out, int out_size, void* d_ws,
                              size_t ws_size, hipStream_t stream) {
  const float* x      = (const float*)d_in[0];
  const float* qkv_u  = (const float*)d_in[1];
  const float* qkv_v  = (const float*)d_in[2];
  const float* proj_u = (const float*)d_in[3];
  const float* proj_v = (const float*)d_in[4];

  char* ws = (char*)d_ws;
  __bf16* Ut  = (__bf16*)(ws + 0);
  __bf16* Vt  = (__bf16*)(ws + 8192);
  __bf16* Put = (__bf16*)(ws + 32768);
  __bf16* Pvt = (__bf16*)(ws + 40960);
  size_t off = 65536;
  const size_t SZ = (size_t)2 * 16 * 2048 * 256 * sizeof(__bf16);  // 32 MB
  __bf16* Qs = (__bf16*)(ws + off); off += SZ;
  __bf16* Ks = (__bf16*)(ws + off); off += SZ;
  __bf16* Vs = (__bf16*)(ws + off); off += SZ;
  __bf16* Yt = (__bf16*)(ws + off); off += SZ;

  k0_prep<<<16, 256, 0, stream>>>(qkv_u, qkv_v, proj_u, proj_v, Ut, Vt, Put, Pvt);
  k1_qkv<<<2048, 64, 0, stream>>>(x, Ut, Vt, Qs, Ks, Vs);
  k2_attn<<<dim3(16, 32, 1), 256, 0, stream>>>(Qs, Ks, Vs, Yt);
  k3_proj<<<2048, 64, 0, stream>>>(Yt, Put, Pvt, (float*)d_out);
}
